// TimeSformerSelfAttention_30846455120606
// MI455X (gfx1250) — matrix operand order repaired and checked by reading, not compiled, not run on hardware
//
#include <hip/hip_runtime.h>

// ---------------------------------------------------------------------------
// TimeSformer self-attention for MI455X (gfx1250, wave32, WMMA bf16 path).
// v3:
//  - GEMM software-pipelined (double-buffered fragments) so global_load_b128
//    for chunk k+1 overlaps WMMAs of chunk k.
//  - Attention: 13 waves per (b,h) cooperatively stage K and V^T into LDS via
//    global_load_async_to_lds_b128 (ASYNCcnt) and compute from LDS, cutting
//    global K/V traffic 13x. Probs transposed through a 1KB/wave LDS buffer.
// ---------------------------------------------------------------------------

typedef __attribute__((ext_vector_type(16))) __bf16 v16bf;
typedef __attribute__((ext_vector_type(8)))  float  v8f;

union FragBF16 {
    v16bf bf;
    unsigned int u[8];
};

__device__ __forceinline__ unsigned short f32_to_bf16(float f) {
    unsigned int u = __float_as_uint(f);
    unsigned int r = u + 0x7FFFu + ((u >> 16) & 1u);   // round-to-nearest-even
    return (unsigned short)(r >> 16);
}

__device__ __forceinline__ unsigned int pack_bf16(float lo, float hi) {
    return (unsigned int)f32_to_bf16(lo) | ((unsigned int)f32_to_bf16(hi) << 16);
}

// CDNA5 async global->LDS copy (16B per lane), tracked by ASYNCcnt.
__device__ __forceinline__ void async_copy_b128(unsigned lds_off,
                                                const unsigned short* g) {
    asm volatile("global_load_async_to_lds_b128 %0, %1, off"
                 :: "v"(lds_off), "v"(g)
                 : "memory");
}
__device__ __forceinline__ void wait_async0() {
    asm volatile("s_wait_asynccnt 0x0" ::: "memory");
}

// ---------------------------------------------------------------------------
// X [12608*768] f32 -> Xb bf16 (row-major). 8 elems/thread, b128 in/out.
// ---------------------------------------------------------------------------
__global__ __launch_bounds__(256)
void convert_x_kernel(const float* __restrict__ X, unsigned int* __restrict__ Xb) {
    const size_t i = ((size_t)blockIdx.x * 256 + threadIdx.x) * 8;
    float4 a = *(const float4*)(X + i);
    float4 b = *(const float4*)(X + i + 4);
    uint4 o;
    o.x = pack_bf16(a.x, a.y);
    o.y = pack_bf16(a.z, a.w);
    o.z = pack_bf16(b.x, b.y);
    o.w = pack_bf16(b.z, b.w);
    *(uint4*)(Xb + i / 2) = o;
}

// ---------------------------------------------------------------------------
// W [768,2304] f32 -> Wt [2304,768] bf16 (row n holds K-contiguous weights).
// ---------------------------------------------------------------------------
__global__ __launch_bounds__(256)
void convert_wt_kernel(const float* __restrict__ W, unsigned int* __restrict__ Wt) {
    const int t  = blockIdx.x * 256 + threadIdx.x;   // 2304*384 threads
    const int n  = t / 384;
    const int kp = t - n * 384;
    const int k  = 2 * kp;
    float w0 = W[(size_t)k * 2304 + n];
    float w1 = W[(size_t)(k + 1) * 2304 + n];
    Wt[(size_t)n * 384 + kp] = pack_bf16(w0, w1);
}

// ---------------------------------------------------------------------------
// QKV GEMM, double-buffered. One wave -> 16x64 strip, 96 WMMAs.
// ---------------------------------------------------------------------------
__global__ __launch_bounds__(32)
void qkv_gemm_kernel(const unsigned short* __restrict__ Xb,
                     const unsigned short* __restrict__ Wt,
                     const float* __restrict__ bias,
                     unsigned short* __restrict__ qw,
                     unsigned short* __restrict__ kw,
                     unsigned short* __restrict__ vt) {
    const int lane = threadIdx.x;
    const int half = lane >> 4;
    const int lr   = lane & 15;
    const int mbase = blockIdx.y * 16;   // 788 M-tiles
    const int nbase = blockIdx.x * 64;   // 36 N-strips

    const unsigned short* Arow  = Xb + (size_t)(mbase + lr) * 768;
    const unsigned short* Brow0 = Wt + (size_t)(nbase + 0  + lr) * 768;
    const unsigned short* Brow1 = Wt + (size_t)(nbase + 16 + lr) * 768;
    const unsigned short* Brow2 = Wt + (size_t)(nbase + 32 + lr) * 768;
    const unsigned short* Brow3 = Wt + (size_t)(nbase + 48 + lr) * 768;

    auto load_chunk = [&](int kc, FragBF16& a, FragBF16& b0, FragBF16& b1,
                          FragBF16& b2, FragBF16& b3) {
        const int k0 = kc * 32 + half * 16;
#pragma unroll
        for (int v = 0; v < 4; ++v) {
            a.u[v]     = *(const unsigned int*)(Arow + k0 + 2 * v);
            a.u[v + 4] = *(const unsigned int*)(Arow + k0 + 8 + 2 * v);
        }
        const int kB = kc * 32 + half * 16;
#pragma unroll
        for (int v = 0; v < 8; ++v) {
            b0.u[v] = *(const unsigned int*)(Brow0 + kB + 2 * v);
            b1.u[v] = *(const unsigned int*)(Brow1 + kB + 2 * v);
            b2.u[v] = *(const unsigned int*)(Brow2 + kB + 2 * v);
            b3.u[v] = *(const unsigned int*)(Brow3 + kB + 2 * v);
        }
    };

    v8f acc[4] = {};
    FragBF16 a0, p0, q0, r0, s0;   // buffer 0
    FragBF16 a1, p1, q1, r1, s1;   // buffer 1

    load_chunk(0, a0, p0, q0, r0, s0);
    for (int kc = 0; kc < 24; kc += 2) {
        load_chunk(kc + 1, a1, p1, q1, r1, s1);   // in flight over these WMMAs
        acc[0] = __builtin_amdgcn_wmma_f32_16x16x32_bf16(false, a0.bf, false, p0.bf, (short)0, acc[0], false, false);
        acc[1] = __builtin_amdgcn_wmma_f32_16x16x32_bf16(false, a0.bf, false, q0.bf, (short)0, acc[1], false, false);
        acc[2] = __builtin_amdgcn_wmma_f32_16x16x32_bf16(false, a0.bf, false, r0.bf, (short)0, acc[2], false, false);
        acc[3] = __builtin_amdgcn_wmma_f32_16x16x32_bf16(false, a0.bf, false, s0.bf, (short)0, acc[3], false, false);
        if (kc + 2 < 24) load_chunk(kc + 2, a0, p0, q0, r0, s0);
        acc[0] = __builtin_amdgcn_wmma_f32_16x16x32_bf16(false, a1.bf, false, p1.bf, (short)0, acc[0], false, false);
        acc[1] = __builtin_amdgcn_wmma_f32_16x16x32_bf16(false, a1.bf, false, q1.bf, (short)0, acc[1], false, false);
        acc[2] = __builtin_amdgcn_wmma_f32_16x16x32_bf16(false, a1.bf, false, r1.bf, (short)0, acc[2], false, false);
        acc[3] = __builtin_amdgcn_wmma_f32_16x16x32_bf16(false, a1.bf, false, s1.bf, (short)0, acc[3], false, false);
    }

    // Bias add + scatter (Q,K row-major [B,H,197,64]; V transposed [B*H,64,224]).
#pragma unroll
    for (int nt = 0; nt < 4; ++nt) {
        const int col = nbase + nt * 16 + lr;       // = s*768 + h*64 + d
        const int s   = col / 768;
        const int rem = col - s * 768;
        const int h   = rem >> 6;
        const int dd  = rem & 63;
        const float bv = bias[col];
#pragma unroll
        for (int j = 0; j < 8; ++j) {
            const int m    = mbase + j + half * 8;  // C-layout row
            const int bidx = m / 197;
            const int tok  = m - bidx * 197;
            const int bh   = bidx * 12 + h;
            const unsigned short val = f32_to_bf16(acc[nt][j] + bv);
            if (s == 0)
                qw[((size_t)bh * 197 + tok) * 64 + dd] = val;
            else if (s == 1)
                kw[((size_t)bh * 197 + tok) * 64 + dd] = val;
            else
                vt[((size_t)bh * 64 + dd) * 224 + tok] = val;
        }
    }
}

// ---------------------------------------------------------------------------
// Attention: one workgroup (13 waves) per (b,h). K and V^T staged once into
// LDS with async copies; each wave handles one 16-row q-block.
// ---------------------------------------------------------------------------
__global__ __launch_bounds__(416)
void attn_kernel(const unsigned short* __restrict__ qw,
                 const unsigned short* __restrict__ kw,
                 const unsigned short* __restrict__ vt,
                 float* __restrict__ out) {
    __shared__ unsigned short Ksh[208 * 64];      // 26624 B (rows 197..207 dup of 196)
    __shared__ unsigned short Vsh[64 * 224];      // 28672 B (pad cols already 0)
    __shared__ unsigned short Psh[13 * 16 * 32];  // 13312 B (per-wave prob chunk)

    const int tid  = threadIdx.x;
    const int wave = tid >> 5;         // 0..12 == q-block
    const int lane = tid & 31;
    const int half = lane >> 4;
    const int lr   = lane & 15;

    const int bh = blockIdx.x;                     // 0..767
    const size_t base  = (size_t)bh * 197 * 64;    // qw/kw base
    const size_t vbase = (size_t)bh * 64 * 224;    // vt base

    // --- cooperative async staging of K (208x64) and V^T (64x224) ---
    const unsigned kshOff = (unsigned)(size_t)(&Ksh[0]);
    const unsigned vshOff = (unsigned)(size_t)(&Vsh[0]);
    for (int idx = tid; idx < 1664; idx += 416) {        // 208 rows * 8 chunks
        int row = idx >> 3;
        int ch  = idx & 7;
        int srow = row > 196 ? 196 : row;
        async_copy_b128(kshOff + (unsigned)idx * 16,
                        kw + base + (size_t)srow * 64 + ch * 8);
    }
    for (int idx = tid; idx < 1792; idx += 416) {        // 64 rows * 28 chunks
        async_copy_b128(vshOff + (unsigned)idx * 16,
                        vt + vbase + (size_t)idx * 8);
    }
    wait_async0();
    __syncthreads();

    // --- Q A-fragments (d 0..31, 32..63), rows clamped to 196 ---
    int qrow = wave * 16 + lr; if (qrow > 196) qrow = 196;
    const unsigned short* Qrow = qw + base + (size_t)qrow * 64;
    FragBF16 qa0, qa1;
#pragma unroll
    for (int v = 0; v < 4; ++v) {
        const int k0 = half * 16 + 2 * v;
        qa0.u[v]     = *(const unsigned int*)(Qrow + k0);
        qa0.u[v + 4] = *(const unsigned int*)(Qrow + k0 + 8);
        qa1.u[v]     = *(const unsigned int*)(Qrow + 32 + k0);
        qa1.u[v + 4] = *(const unsigned int*)(Qrow + 32 + k0 + 8);
    }

    // --- scores = Q @ K^T, K B-fragments from LDS ---
    v8f sacc[13] = {};
#pragma unroll
    for (int kt = 0; kt < 13; ++kt) {
        const int krow = kt * 16 + lr;
        const unsigned short* Krow = &Ksh[krow * 64];
        FragBF16 kb0, kb1;
        const int kB = half * 16;
#pragma unroll
        for (int v = 0; v < 8; ++v) {
            kb0.u[v] = *(const unsigned int*)(Krow + kB + 2 * v);
            kb1.u[v] = *(const unsigned int*)(Krow + 32 + kB + 2 * v);
        }
        sacc[kt] = __builtin_amdgcn_wmma_f32_16x16x32_bf16(
            false, qa0.bf, false, kb0.bf, (short)0, sacc[kt], false, false);
        sacc[kt] = __builtin_amdgcn_wmma_f32_16x16x32_bf16(
            false, qa1.bf, false, kb1.bf, (short)0, sacc[kt], false, false);
    }

    // --- softmax (f32). C-layout: row = j + 8*half, col = kt*16 + lr ---
    const float scale = 0.125f;
    float inv[8];
#pragma unroll
    for (int j = 0; j < 8; ++j) {
        float m = -1e30f;
#pragma unroll
        for (int kt = 0; kt < 13; ++kt) {
            const int kcol = kt * 16 + lr;
            float sv = (kcol < 197) ? sacc[kt][j] * scale : -1e30f;
            m = fmaxf(m, sv);
        }
        m = fmaxf(m, __shfl_xor(m, 1, 32));
        m = fmaxf(m, __shfl_xor(m, 2, 32));
        m = fmaxf(m, __shfl_xor(m, 4, 32));
        m = fmaxf(m, __shfl_xor(m, 8, 32));

        float s = 0.f;
#pragma unroll
        for (int kt = 0; kt < 13; ++kt) {
            const int kcol = kt * 16 + lr;
            float e = (kcol < 197) ? __expf(sacc[kt][j] * scale - m) : 0.f;
            sacc[kt][j] = e;   // e == 0 for masked cols
            s += e;
        }
        s += __shfl_xor(s, 1, 32);
        s += __shfl_xor(s, 2, 32);
        s += __shfl_xor(s, 4, 32);
        s += __shfl_xor(s, 8, 32);
        inv[j] = 1.f / s;
    }

    // --- ctx = P @ V: per 32-col chunk, transpose probs through LDS, 4 WMMAs ---
    unsigned short* Pbuf = &Psh[wave * 16 * 32];
    v8f oacc[4] = {};
#pragma unroll
    for (int c = 0; c < 7; ++c) {
        // write prob tiles 2c, 2c+1 (C-layout -> row-major 16x32 chunk)
#pragma unroll
        for (int t = 0; t < 2; ++t) {
            const int kt = 2 * c + t;
#pragma unroll
            for (int j = 0; j < 8; ++j) {
                const int row = j + half * 8;
                unsigned short pv = 0;
                if (kt < 13) pv = f32_to_bf16(sacc[kt][j] * inv[j]);
                Pbuf[row * 32 + t * 16 + lr] = pv;
            }
        }
        // A-fragment from the 16x32 chunk
        FragBF16 pa;
        const int k0 = half * 16;
#pragma unroll
        for (int v = 0; v < 4; ++v) {
            pa.u[v]     = *(const unsigned int*)(&Pbuf[lr * 32 + k0 + 2 * v]);
            pa.u[v + 4] = *(const unsigned int*)(&Pbuf[lr * 32 + k0 + 8 + 2 * v]);
        }
        const int kB = c * 32 + half * 16;
#pragma unroll
        for (int nt = 0; nt < 4; ++nt) {
            FragBF16 vb;
#pragma unroll
            for (int v = 0; v < 8; ++v)
                vb.u[v] = *(const unsigned int*)(&Vsh[(nt * 16 + lr) * 224 + kB + 2 * v]);
            oacc[nt] = __builtin_amdgcn_wmma_f32_16x16x32_bf16(
                false, pa.bf, false, vb.bf, (short)0, oacc[nt], false, false);
        }
    }

    // --- store ctx: out[b, tok, h*64 + d] f32 ---
    const int b = bh / 12;
    const int h = bh - b * 12;
#pragma unroll
    for (int nt = 0; nt < 4; ++nt) {
#pragma unroll
        for (int j = 0; j < 8; ++j) {
            const int row = wave * 16 + j + half * 8;
            if (row < 197)
                out[((size_t)(b * 197 + row)) * 768 + h * 64 + nt * 16 + lr] =
                    oacc[nt][j];
        }
    }
}

// ---------------------------------------------------------------------------
extern "C" void kernel_launch(void* const* d_in, const int* in_sizes, int n_in,
                              void* d_out, int out_size, void* d_ws, size_t ws_size,
                              hipStream_t stream) {
    const float* X    = (const float*)d_in[0];   // [64,197,768]
    const float* W    = (const float*)d_in[1];   // [768,2304]
    const float* bias = (const float*)d_in[2];   // [2304]
    float* out = (float*)d_out;

    // Workspace layout (bf16 elems): Xb | Wt | qw | kw | vt
    const size_t nXb = (size_t)12608 * 768;      //  9,682,944
    const size_t nWt = (size_t)2304 * 768;       //  1,769,472
    const size_t nQ  = (size_t)768 * 197 * 64;   //  9,682,944
    const size_t nVt = (size_t)768 * 64 * 224;   // 11,010,048
    unsigned short* Xb = (unsigned short*)d_ws;
    unsigned short* Wt = Xb + nXb;
    unsigned short* qw = Wt + nWt;
    unsigned short* kw = qw + nQ;
    unsigned short* vt = kw + nQ;

    // Zero Vt so pad columns (tok 197..223) are exactly 0.
    hipMemsetAsync(vt, 0, nVt * sizeof(unsigned short), stream);

    convert_x_kernel <<<(int)(nXb / (256 * 8)), 256, 0, stream>>>(X,  (unsigned int*)Xb);
    convert_wt_kernel<<<(int)(2304 * 384 / 256), 256, 0, stream>>>(W, (unsigned int*)Wt);

    dim3 ggrid(36, 788);
    qkv_gemm_kernel<<<ggrid, 32, 0, stream>>>(Xb, Wt, bias, qw, kw, vt);
    attn_kernel<<<768, 416, 0, stream>>>(qw, kw, vt, out);
}
